// Atten_28535762715239
// MI455X (gfx1250) — matrix operand order repaired and checked by reading, not compiled, not run on hardware
//
#include <hip/hip_runtime.h>
#include <hip/hip_bf16.h>

// ---- sizes from the reference ----
#define BB   4
#define SS   2048
#define HID_ 1024

typedef __attribute__((ext_vector_type(16))) __bf16 v16bf;
typedef __attribute__((ext_vector_type(8)))  __bf16 v8bf;
typedef __attribute__((ext_vector_type(4)))  __bf16 v4bf;
typedef __attribute__((ext_vector_type(8)))  float  v8f;

// ---------------------------------------------------------------------------
// CDNA5 async global->LDS copy (16B per lane) + ASYNCcnt waits
// ---------------------------------------------------------------------------
__device__ __forceinline__ void async_b128(void* lds_ptr, const void* gptr) {
    // generic LDS address: low 32 bits are the LDS offset
    unsigned lds_off = (unsigned)(size_t)lds_ptr;
    asm volatile("global_load_async_to_lds_b128 %0, %1, off"
                 :: "v"(lds_off), "v"(gptr) : "memory");
}

#if defined(__has_builtin) && __has_builtin(__builtin_amdgcn_s_wait_asynccnt)
#define WAIT_ASYNC_4() __builtin_amdgcn_s_wait_asynccnt(4)
#define WAIT_ASYNC_0() __builtin_amdgcn_s_wait_asynccnt(0)
#else
#define WAIT_ASYNC_4() asm volatile("s_wait_asynccnt 0x4" ::: "memory")
#define WAIT_ASYNC_0() asm volatile("s_wait_asynccnt 0x0" ::: "memory")
#endif

// ---------------------------------------------------------------------------
// fp32 -> bf16 bulk convert (vectorized by 4)
// ---------------------------------------------------------------------------
__global__ void cvt_f32_bf16(const float* __restrict__ s, __bf16* __restrict__ d,
                             long long n4) {
    long long i = (long long)blockIdx.x * blockDim.x + threadIdx.x;
    long long stride = (long long)gridDim.x * blockDim.x;
    for (; i < n4; i += stride) {
        float4 f = ((const float4*)s)[i];
        v4bf o = { (__bf16)f.x, (__bf16)f.y, (__bf16)f.z, (__bf16)f.w };
        ((v4bf*)d)[i] = o;
    }
}

// ---------------------------------------------------------------------------
// WMMA fragment loaders (bf16, 16x16x32)
// A 16x32: lane L holds row M=L%16; K = {h*8..h*8+7, 16+h*8..16+h*8+7}, h=L/16
// B 32x16 (loaded from Bt rows, Bt[n][k]): lane L holds col N=L%16;
//          K = {h*8..h*8+7, 16+h*8..16+h*8+7} (same as A)
// ---------------------------------------------------------------------------
__device__ __forceinline__ v16bf load_fragA(const __bf16* base, int ld, int k0,
                                            int lane) {
    int r = lane & 15, h = lane >> 4;
    const __bf16* p = base + (long long)r * ld + k0 + h * 8;
    union { v16bf v; v8bf q[2]; } u;
    u.q[0] = *(const v8bf*)(p);
    u.q[1] = *(const v8bf*)(p + 16);
    return u.v;
}

__device__ __forceinline__ v16bf load_fragB(const __bf16* btrow0, int ld, int k0,
                                            int lane) {
    int n = lane & 15, h = lane >> 4;
    const __bf16* p = btrow0 + (long long)n * ld + k0 + h * 8;
    union { v16bf v; v8bf q[2]; } u;
    u.q[0] = *(const v8bf*)(p);
    u.q[1] = *(const v8bf*)(p + 16);
    return u.v;
}

union AccU { v8f v; float f[8]; };

// ---------------------------------------------------------------------------
// Generic batched GEMM:  C[b] = act(A[b] (MxK) * Bt[b](NxK)^T + bias)
// 256 threads = 8 waves (4 M x 2 N); wave tile 32x64 (2x4 WMMA tiles).
// Block tile 128(M) x 128(N).  K stepped by 32 with LDS double buffering:
// A-tile 128x32 + Bt-tile 128x32 staged via global_load_async_to_lds_b128.
// TRANS: store C transposed ([N][ldc], col index m).
// ---------------------------------------------------------------------------
template <bool RELU, bool TRANS, typename OUT_T>
__global__ __launch_bounds__(256)
void gemm_bf16_wmma(const __bf16* __restrict__ A, const __bf16* __restrict__ Bt,
                    const float* __restrict__ bias, OUT_T* __restrict__ C,
                    int M, int N, int K,
                    long long sA, long long sB, long long sC, int ldc) {
    __shared__ __align__(16) __bf16 sa[2][128 * 32];   // 8 KB x2
    __shared__ __align__(16) __bf16 sb[2][128 * 32];   // 8 KB x2

    int b = blockIdx.z;
    A  += (long long)b * sA;
    Bt += (long long)b * sB;
    C  += (long long)b * sC;

    const int tid  = threadIdx.x;
    const int lane = tid & 31;
    const int wid  = tid >> 5;
    const int wm   = wid >> 1;            // 0..3
    const int wn   = wid & 1;             // 0..1
    const int mblk = blockIdx.y * 128;
    const int nblk = blockIdx.x * 128;

    const __bf16* Ab = A  + (long long)mblk * K;
    const __bf16* Bb = Bt + (long long)nblk * K;

    // stage copier: 128 rows x 32 bf16 (64B) per tile -> 512 16B units each
    auto stage = [&](int k0, int buf) {
        for (int u = tid; u < 512; u += 256) {
            int row = u >> 2, ch = (u & 3) * 8;
            async_b128(&sa[buf][row * 32 + ch], Ab + (long long)row * K + k0 + ch);
            async_b128(&sb[buf][row * 32 + ch], Bb + (long long)row * K + k0 + ch);
        }
    };

    AccU acc[2][4] = {};
    const int nk = K / 32;

    stage(0, 0);
    for (int kk = 0; kk < nk; ++kk) {
        int buf = kk & 1;
        if (kk + 1 < nk) { stage((kk + 1) * 32, buf ^ 1); WAIT_ASYNC_4(); }
        else             { WAIT_ASYNC_0(); }
        __syncthreads();

        v16bf a0 = load_fragA(&sa[buf][(wm * 32 +  0) * 32], 32, 0, lane);
        v16bf a1 = load_fragA(&sa[buf][(wm * 32 + 16) * 32], 32, 0, lane);
        for (int j = 0; j < 4; ++j) {
            v16bf bf = load_fragB(&sb[buf][(wn * 64 + j * 16) * 32], 32, 0, lane);
            acc[0][j].v = __builtin_amdgcn_wmma_f32_16x16x32_bf16(
                false, a0, false, bf, (short)0, acc[0][j].v, false, false);
            acc[1][j].v = __builtin_amdgcn_wmma_f32_16x16x32_bf16(
                false, a1, false, bf, (short)0, acc[1][j].v, false, false);
        }
        __syncthreads();   // all waves done reading buf before it is re-staged
    }

    const int h = lane >> 4;
    for (int i = 0; i < 2; ++i) {
        for (int j = 0; j < 4; ++j) {
            int n  = nblk + wn * 64 + j * 16 + (lane & 15);
            float bv = bias ? bias[n] : 0.0f;
            for (int r = 0; r < 8; ++r) {
                int m = mblk + wm * 32 + i * 16 + r + 8 * h;
                float val = acc[i][j].f[r] + bv;
                if (RELU) val = fmaxf(val, 0.0f);
                if (TRANS) C[(long long)n * ldc + m] = (OUT_T)val;
                else       C[(long long)m * ldc + n] = (OUT_T)val;
            }
        }
    }
}

// ---------------------------------------------------------------------------
// Attention scores + softmax.
// One workgroup per (batch, 16-query-row block). 8 waves; wave w owns key
// columns [w*256, w*256+256) -> 16 WMMA tiles held in registers.
// Q rows staged in LDS via async copy; softmax reduced wave-local via
// shuffles then cross-wave via LDS.
// ---------------------------------------------------------------------------
__global__ __launch_bounds__(256)
void attn_softmax_wmma(const __bf16* __restrict__ Qb, const __bf16* __restrict__ Kb,
                       float* __restrict__ attn_f, __bf16* __restrict__ attn_b,
                       float scale) {
    const int S = SS, D = HID_;
    int b    = blockIdx.y;
    int qblk = blockIdx.x;                       // 16-row query block

    __shared__ __align__(16) __bf16 qs[16 * HID_];   // 32 KB
    __shared__ float  wpart[8][16];
    __shared__ float  rowred[16];

    const __bf16* Qrow = Qb + ((long long)b * S + (long long)qblk * 16) * D;
    const __bf16* Kbb  = Kb + (long long)b * S * D;

    // async 32KB copy: 2048 x 16B units, 8 per thread
    for (int u = threadIdx.x; u < 2048; u += 256)
        async_b128(qs + u * 8, Qrow + u * 8);
    WAIT_ASYNC_0();
    __syncthreads();

    int lane = threadIdx.x & 31;
    int wid  = threadIdx.x >> 5;
    int ncol0 = wid * 256;
    int h = lane >> 4;

    AccU acc[16] = {};
    for (int k0 = 0; k0 < D; k0 += 32) {
        v16bf a = load_fragA(qs, D, k0, lane);
        for (int t = 0; t < 16; ++t) {
            v16bf bf = load_fragB(Kbb + (long long)(ncol0 + t * 16) * D, D, k0, lane);
            acc[t].v = __builtin_amdgcn_wmma_f32_16x16x32_bf16(
                false, a, false, bf, (short)0, acc[t].v, false, false);
        }
    }

    // ---- row max (rows owned by this lane: h*8 + r, r=0..7) ----
    float rmax[8];
    for (int r = 0; r < 8; ++r) {
        float m = -3.0e38f;
        for (int t = 0; t < 16; ++t) m = fmaxf(m, acc[t].f[r]);
        for (int off = 1; off < 16; off <<= 1)
            m = fmaxf(m, __shfl_xor(m, off, 16));
        rmax[r] = m;
    }
    if ((lane & 15) == 0)
        for (int r = 0; r < 8; ++r) wpart[wid][h * 8 + r] = rmax[r];
    __syncthreads();
    if (threadIdx.x < 16) {
        float m = wpart[0][threadIdx.x];
        for (int w = 1; w < 8; ++w) m = fmaxf(m, wpart[w][threadIdx.x]);
        rowred[threadIdx.x] = m;
    }
    __syncthreads();

    // ---- exponentiate + row sum ----
    float rsum[8];
    for (int r = 0; r < 8; ++r) {
        float m = rowred[h * 8 + r];
        float s = 0.0f;
        for (int t = 0; t < 16; ++t) {
            float p = __expf((acc[t].f[r] - m) * scale);
            acc[t].f[r] = p;
            s += p;
        }
        for (int off = 1; off < 16; off <<= 1)
            s += __shfl_xor(s, off, 16);
        rsum[r] = s;
    }
    __syncthreads();                              // done reading rowred/wpart
    if ((lane & 15) == 0)
        for (int r = 0; r < 8; ++r) wpart[wid][h * 8 + r] = rsum[r];
    __syncthreads();
    if (threadIdx.x < 16) {
        float s = 0.0f;
        for (int w = 0; w < 8; ++w) s += wpart[w][threadIdx.x];
        rowred[threadIdx.x] = s;
    }
    __syncthreads();

    // ---- normalize + store (fp32 to d_out, bf16 to workspace) ----
    for (int r = 0; r < 8; ++r) {
        int row = qblk * 16 + h * 8 + r;
        float inv = 1.0f / rowred[h * 8 + r];
        long long rowbase = ((long long)b * S + row) * S;
        for (int t = 0; t < 16; ++t) {
            int col = ncol0 + t * 16 + (lane & 15);
            float p = acc[t].f[r] * inv;
            attn_f[rowbase + col] = p;
            attn_b[rowbase + col] = (__bf16)p;
        }
    }
}

// ---------------------------------------------------------------------------
// launch
// ---------------------------------------------------------------------------
extern "C" void kernel_launch(void* const* d_in, const int* in_sizes, int n_in,
                              void* d_out, int out_size, void* d_ws, size_t ws_size,
                              hipStream_t stream) {
    (void)in_sizes; (void)n_in; (void)out_size; (void)ws_size;

    const float* q     = (const float*)d_in[0];
    const float* k     = (const float*)d_in[1];
    const float* v     = (const float*)d_in[2];
    const float* wQ_W  = (const float*)d_in[3];
    const float* wQ_b  = (const float*)d_in[4];
    const float* wK_W  = (const float*)d_in[5];
    const float* wK_b  = (const float*)d_in[6];
    // d_in[7], d_in[8]: wV_W / wV_b are dead (reference reuses wQ for V)
    const float* out_W = (const float*)d_in[9];
    const float* out_b = (const float*)d_in[10];

    const long long NBS  = (long long)BB * SS * HID_;   // 8388608
    const long long NW   = (long long)HID_ * HID_;      // 1048576
    const long long NATT = (long long)BB * SS * SS;     // 16777216

    __bf16* wsb  = (__bf16*)d_ws;
    __bf16* qb   = wsb;                 // [B,S,H]
    __bf16* kb   = qb  + NBS;
    __bf16* vb   = kb  + NBS;
    __bf16* wqb  = vb  + NBS;           // [H,H]
    __bf16* wkb  = wqb + NW;
    __bf16* owb  = wkb + NW;
    __bf16* Qb   = owb + NW;            // [B,S,H]
    __bf16* Kb   = Qb  + NBS;
    __bf16* Vtb  = Kb  + NBS;           // [B,H,S]  (V transposed)
    __bf16* attb = Vtb + NBS;           // [B,S,S]
    __bf16* ctxb = attb + NATT;         // [B,S,H]

    float* out_f  = (float*)d_out;          // [B,S,H]
    float* attn_f = out_f + NBS;            // [B,S,S]

    // 1) fp32 -> bf16 conversions
    dim3 cb(256), cg(2048);
    cvt_f32_bf16<<<cg, cb, 0, stream>>>(q,     qb,  NBS / 4);
    cvt_f32_bf16<<<cg, cb, 0, stream>>>(k,     kb,  NBS / 4);
    cvt_f32_bf16<<<cg, cb, 0, stream>>>(v,     vb,  NBS / 4);
    cvt_f32_bf16<<<cg, cb, 0, stream>>>(wQ_W,  wqb, NW / 4);
    cvt_f32_bf16<<<cg, cb, 0, stream>>>(wK_W,  wkb, NW / 4);
    cvt_f32_bf16<<<cg, cb, 0, stream>>>(out_W, owb, NW / 4);

    const long long sAct = (long long)SS * HID_;   // per-batch activation stride
    dim3 gblk(256);
    dim3 gproj(HID_ / 128, SS / 128, BB);          // (8, 16, 4)

    // 2) projections: Q = relu(q wQ^T), K = relu(k wK^T), V^T = relu(v wQ^T)^T
    gemm_bf16_wmma<true, false, __bf16><<<gproj, gblk, 0, stream>>>(
        qb, wqb, wQ_b, Qb, SS, HID_, HID_, sAct, 0, sAct, HID_);
    gemm_bf16_wmma<true, false, __bf16><<<gproj, gblk, 0, stream>>>(
        kb, wkb, wK_b, Kb, SS, HID_, HID_, sAct, 0, sAct, HID_);
    gemm_bf16_wmma<true, true, __bf16><<<gproj, gblk, 0, stream>>>(
        vb, wqb, wQ_b, Vtb, SS, HID_, HID_, sAct, 0, sAct, SS);

    // 3) scores + softmax  (scale = 1/sqrt(1024) = 0.03125)
    dim3 ag(SS / 16, BB);
    attn_softmax_wmma<<<ag, dim3(256), 0, stream>>>(Qb, Kb, attn_f, attb, 0.03125f);

    // 4) ctx = attn @ V   (A=[S,S] bf16, Bt=V^T rows = columns of V)
    dim3 gctx(HID_ / 128, SS / 128, BB);
    gemm_bf16_wmma<false, false, __bf16><<<gctx, gblk, 0, stream>>>(
        attb, Vtb, nullptr, ctxb, SS, HID_, SS,
        (long long)SS * SS, sAct, sAct, HID_);

    // 5) out = ctx @ out_W^T + out_b  (fp32 output)
    gemm_bf16_wmma<false, false, float><<<gproj, gblk, 0, stream>>>(
        ctxb, owb, out_b, out_f, SS, HID_, HID_, sAct, 0, sAct, HID_);
}
